// GraphAttentionLayer_20916490732047
// MI455X (gfx1250) — matrix operand order repaired and checked by reading, not compiled, not run on hardware
//
#include <hip/hip_runtime.h>
#include <hip/hip_bf16.h>

// GAT layer, N=4096, IN_F=512, H=8, F=32.
// K1: x = NR @ W via v_wmma_f32_16x16x4_f32 (f32-exact), emit x_f32 + xT_f16[h][f][j]
// K2: s_src/s_tgt per (node, head) + per-head max(s_src) via ordered-uint atomicMax
// K3: flash-style fused masked-softmax + (att @ x) via v_wmma_f32_16x16x32_f16,
//     P-tile built directly in A-fragment registers, row sums via shfl_xor(16).

typedef __attribute__((ext_vector_type(16))) _Float16 v16h;
typedef __attribute__((ext_vector_type(8)))  _Float16 v8h;
typedef __attribute__((ext_vector_type(8)))  float    v8f;
typedef __attribute__((ext_vector_type(2)))  float    v2f;

#define N_NODES 4096
#define IN_F    512
#define NHEADS  8
#define FDIM    32
#define OUTW    256   // H*F

// workspace byte offsets
#define OFF_X32  0u                      // 4096*256 f32  = 4 MiB
#define OFF_XT   4194304u                // 8*32*4096 f16 = 2 MiB
#define OFF_SS   6291456u                // 8*4096 f32    = 128 KiB
#define OFF_ST   6422528u                // 8*4096 f32    = 128 KiB
#define OFF_M    6553600u                // 8 u32

__device__ __forceinline__ unsigned int fenc(float f) {
  unsigned int b = __float_as_uint(f);
  return (b & 0x80000000u) ? ~b : (b | 0x80000000u);
}
__device__ __forceinline__ float fdec(unsigned int u) {
  unsigned int b = (u & 0x80000000u) ? (u & 0x7FFFFFFFu) : ~u;
  return __uint_as_float(b);
}

__global__ void gat_init_kernel(unsigned int* u_m) {
  if (threadIdx.x < NHEADS) u_m[threadIdx.x] = 0u;  // encoded "below -inf"
}

// ---------------- K1: x = NR @ W (f32 WMMA), one wave per 16x16 C tile ----------------
__global__ void gat_gemm_kernel(const float* __restrict__ NR,
                                const float* __restrict__ W,
                                float* __restrict__ x32,
                                _Float16* __restrict__ xt16) {
  const int wave = threadIdx.x >> 5;
  const int lane = threadIdx.x & 31;
  const int hi = lane >> 4;      // half-wave select
  const int r  = lane & 15;
  const int tile = blockIdx.x * 8 + wave;      // 4096 tiles total
  const int i0 = (tile >> 4) * 16;             // row tile (0..255)*16
  const int j0 = (tile & 15) * 16;             // col tile (0..15)*16

  const float* arow = NR + (size_t)(i0 + r) * IN_F;
  v8f acc = {};
  for (int k0 = 0; k0 < IN_F; k0 += 4) {
    const int kA = k0 + hi * 2;
    // A 16x4: lanes0-15 K={0,1}, lanes16-31 K={2,3}
    float2 a2 = *(const float2*)(arow + kA);
    v2f A; A[0] = a2.x; A[1] = a2.y;
    // B 4x16: lanes0-15 N=col K={0,1}, lanes16-31 K={2,3}
    v2f B;
    B[0] = W[(size_t)kA * OUTW + j0 + r];
    B[1] = W[(size_t)(kA + 1) * OUTW + j0 + r];
    acc = __builtin_amdgcn_wmma_f32_16x16x4_f32(false, A, false, B,
                                                (short)0, acc, false, false);
  }

  // C layout: lane n holds col j0+(n&15), rows i0 + v + hi*8 (v = 0..7)
  const int h  = j0 >> 5;                 // head of this col tile
  const int col = j0 + r;                 // global feature column
  v8h pk;
#pragma unroll
  for (int v = 0; v < 8; ++v) {
    const size_t row = (size_t)(i0 + v + hi * 8);
    x32[row * OUTW + col] = acc[v];
    pk[v] = (_Float16)acc[v];
  }
  // xT_f16[h][f][j]: lane's 8 rows are contiguous in j -> one 16B store
  *(v8h*)(xt16 + ((size_t)col) * N_NODES + i0 + hi * 8) = pk;
}

// ---------------- K2: per-(node,head) scores + per-head max(s_src) ----------------
__global__ void gat_scores_kernel(const float* __restrict__ x32,
                                  const float* __restrict__ a,
                                  float* __restrict__ sTsrc,
                                  float* __restrict__ sTtgt,
                                  unsigned int* __restrict__ u_m) {
  const int tid = blockIdx.x * blockDim.x + threadIdx.x;   // 32768 total
  const int n = tid >> 3;
  const int h = tid & 7;
  const float* xr = x32 + (size_t)n * OUTW + h * FDIM;
  float ss = 0.f, st = 0.f;
#pragma unroll
  for (int f = 0; f < FDIM; ++f) {
    const float xv = xr[f];
    ss += xv * a[f];          // source contribution
    st += xv * a[FDIM + f];   // target contribution
  }
  sTsrc[(size_t)h * N_NODES + n] = ss;
  sTtgt[(size_t)h * N_NODES + n] = st;
  // lanes {l, l^8, l^16, l^24} share head h = l&7: reduce before atomic
  float mx = ss;
  mx = fmaxf(mx, __shfl_xor(mx, 8, 32));
  mx = fmaxf(mx, __shfl_xor(mx, 16, 32));
  if ((threadIdx.x & 31) < NHEADS) atomicMax(&u_m[h], fenc(mx));
}

// ---------------- K3: fused masked softmax + att@x (flash style) ----------------
// Block = 8 waves = 8 heads; one 16-target-row tile per block.
__global__ void gat_attn_kernel(const unsigned char* __restrict__ adj,
                                const _Float16* __restrict__ xt16,
                                const float* __restrict__ sTsrc,
                                const float* __restrict__ sTtgt,
                                const unsigned int* __restrict__ u_m,
                                float* __restrict__ out) {
  const int h    = threadIdx.x >> 5;
  const int lane = threadIdx.x & 31;
  const int hi = lane >> 4;
  const int r  = lane & 15;
  const int i0 = blockIdx.x * 16;
  const int koff = hi * 16;          // A-fragment K base: lanes0-15 -> 0, 16-31 -> 16

  const float stgt = sTtgt[(size_t)h * N_NODES + i0 + r];
  // per-row safe max: leaky_relu is monotone, so leaky(stgt + max_j ssrc) >= all e
  float mrow = stgt + fdec(u_m[h]);
  mrow = mrow > 0.f ? mrow : 0.2f * mrow;

  const unsigned char* arow = adj + (size_t)(i0 + r) * N_NODES;
  const float* ssrc = sTsrc + (size_t)h * N_NODES;
  const _Float16* b0p = xt16 + ((size_t)(h * FDIM + r)) * N_NODES;       // cols 0..15
  const _Float16* b1p = xt16 + ((size_t)(h * FDIM + 16 + r)) * N_NODES;  // cols 16..31

  v8f acc0 = {}, acc1 = {};
  float rowsum = 0.f;

  for (int j0 = 0; j0 < N_NODES; j0 += 32) {
    // adjacency bytes exactly matching this lane's A-fragment K indices
    const unsigned long long adjA = *(const unsigned long long*)(arow + j0 + koff);
    const unsigned long long adjB = *(const unsigned long long*)(arow + j0 + koff + 8);
    // s_src for the same K indices (uniform across each half-wave -> broadcast)
    alignas(16) float sv[16];
    *(float4*)(sv + 0)  = *(const float4*)(ssrc + j0 + koff);
    *(float4*)(sv + 4)  = *(const float4*)(ssrc + j0 + koff + 4);
    *(float4*)(sv + 8)  = *(const float4*)(ssrc + j0 + koff + 8);
    *(float4*)(sv + 12) = *(const float4*)(ssrc + j0 + koff + 12);

    v16h A;
#pragma unroll
    for (int e = 0; e < 16; ++e) {
      const unsigned int ab =
          (unsigned int)(((e < 8) ? (adjA >> (8 * e)) : (adjB >> (8 * (e - 8)))) & 0xFFull);
      float ev = stgt + sv[e];
      ev = ev > 0.f ? ev : 0.2f * ev;          // leaky_relu(.., 0.2)
      const float p = ab ? __expf(ev - mrow) : 0.f;
      rowsum += p;
      A[e] = (_Float16)p;
    }

    // B fragments: 16 contiguous f16 per lane from xT[h][col][j0 + hi*16 ..]
    const v16h B0 = *(const v16h*)(b0p + j0 + hi * 16);
    const v16h B1 = *(const v16h*)(b1p + j0 + hi * 16);
    acc0 = __builtin_amdgcn_wmma_f32_16x16x32_f16(false, A, false, B0,
                                                  (short)0, acc0, false, false);
    acc1 = __builtin_amdgcn_wmma_f32_16x16x32_f16(false, A, false, B1,
                                                  (short)0, acc1, false, false);
  }

  // lanes l and l^16 hold the two K-halves of the same row r: combine
  const float ell = rowsum + __shfl_xor(rowsum, 16, 32);   // lane now holds L[r]

#pragma unroll
  for (int v = 0; v < 8; ++v) {
    const float linv = 1.0f / __shfl(ell, v + hi * 8, 32); // L for this C row
    const size_t row = (size_t)(i0 + v + hi * 8);
    out[row * OUTW + h * FDIM + r]      = acc0[v] * linv;
    out[row * OUTW + h * FDIM + 16 + r] = acc1[v] * linv;
  }
}

extern "C" void kernel_launch(void* const* d_in, const int* in_sizes, int n_in,
                              void* d_out, int out_size, void* d_ws, size_t ws_size,
                              hipStream_t stream) {
  const float* NR          = (const float*)d_in[0];
  const unsigned char* adj = (const unsigned char*)d_in[1];  // jnp bool = 1 byte
  const float* W           = (const float*)d_in[2];
  const float* a           = (const float*)d_in[3];
  float* out               = (float*)d_out;

  char* ws = (char*)d_ws;
  float*        x32   = (float*)(ws + OFF_X32);
  _Float16*     xt16  = (_Float16*)(ws + OFF_XT);
  float*        sTsrc = (float*)(ws + OFF_SS);
  float*        sTtgt = (float*)(ws + OFF_ST);
  unsigned int* u_m   = (unsigned int*)(ws + OFF_M);

  gat_init_kernel<<<1, 32, 0, stream>>>(u_m);
  gat_gemm_kernel<<<512, 256, 0, stream>>>(NR, W, x32, xt16);     // 4096 tiles / 8 waves
  gat_scores_kernel<<<128, 256, 0, stream>>>(x32, a, sTsrc, sTtgt, u_m);
  gat_attn_kernel<<<256, 256, 0, stream>>>(adj, xt16, sTsrc, sTtgt, u_m, out);
}
